// Decoder_56581899157956
// MI455X (gfx1250) — matrix operand order repaired and checked by reading, not compiled, not run on hardware
//
#include <hip/hip_runtime.h>

#define Bc 2
#define Tc 1024
#define Sc 1024
#define Dc 512
#define Hc 8
#define DHc 64
#define Lc 4
#define Ec 8
#define FFc 2048

typedef __attribute__((ext_vector_type(16))) __bf16 v16bf;
typedef __attribute__((ext_vector_type(8)))  __bf16 v8bf;
typedef __attribute__((ext_vector_type(8)))  float  v8f;

__device__ inline unsigned short f2bf(float f) {
    unsigned u = __builtin_bit_cast(unsigned, f);
    u += 0x7FFFu + ((u >> 16) & 1u);           // round-to-nearest-even
    return (unsigned short)(u >> 16);
}
__device__ inline __bf16 bfraw(unsigned short u) { return __builtin_bit_cast(__bf16, u); }

__device__ inline void out_write(float* p, float v, bool acc) { if (acc) *p += v; else *p = v; }
__device__ inline void out_write(unsigned short* p, float v, bool acc) { (void)acc; *p = f2bf(v); }

// ---------------------------------------------------------------------------
// DPP16 butterfly reductions (no LDS, no waits).
// ---------------------------------------------------------------------------
template<int C>
__device__ inline float dpp_xchg(float x) {
    return __builtin_bit_cast(float, __builtin_amdgcn_update_dpp(
        0, __builtin_bit_cast(int, x), C, 0xf, 0xf, true));
}
__device__ inline float red16_max(float x) {       // uniform across each 16-lane row
    x = fmaxf(x, dpp_xchg<0xB1>(x));               // quad_perm(1,0,3,2)
    x = fmaxf(x, dpp_xchg<0x4E>(x));               // quad_perm(2,3,0,1)
    x = fmaxf(x, dpp_xchg<0x141>(x));              // row_half_mirror
    x = fmaxf(x, dpp_xchg<0x140>(x));              // row_mirror
    return x;
}
__device__ inline float red16_sum(float x) {
    x += dpp_xchg<0xB1>(x);
    x += dpp_xchg<0x4E>(x);
    x += dpp_xchg<0x141>(x);
    x += dpp_xchg<0x140>(x);
    return x;
}
__device__ inline float red32_sum(float x) {       // full-wave sum
    x = red16_sum(x);
    x += __shfl_xor(x, 16, 32);
    return x;
}

// Assemble a 16-elem fragment from two contiguous 8-elem bf16 runs (2 x b128 loads).
__device__ inline v16bf load_frag_2x8(const unsigned short* p0, const unsigned short* p1) {
    v8bf lo = *(const v8bf*)p0;
    v8bf hi = *(const v8bf*)p1;
    v16bf r;
#pragma unroll
    for (int i = 0; i < 8; i++) { r[i] = lo[i]; r[i + 8] = hi[i]; }
    return r;
}

// ---------------------------------------------------------------------------
// Generic GEMM with TRANSPOSED weights:
//   C = op(A[M,K]bf16 @ W + bias) * rowscale,  W given as Wt[N,K] bf16.
// Per-lane base pointers + immediate offsets; one pointer add per K-step.
// One wave computes a 32x64 tile; 12 b128 loads then 8 WMMA per K-step.
// ---------------------------------------------------------------------------
template<bool RELU, bool ACC, bool TROUT, typename OutT>
__global__ __launch_bounds__(256, 1)
void gemm_bf16_wmma(const unsigned short* __restrict__ A,
                    const unsigned short* __restrict__ Wt,
                    const float* __restrict__ bias,
                    const float* __restrict__ rowscale, int rs_stride,
                    OutT* __restrict__ C,
                    int M, int N, int Kd)
{
    const int lane = threadIdx.x & 31;
    const int hi   = lane >> 4;      // lane half
    const int lc   = lane & 15;
    const int tiles_n = N >> 6;
    const int tile  = blockIdx.x * (blockDim.x >> 5) + (threadIdx.x >> 5);
    const int total = (M >> 5) * tiles_n;
    if (tile >= total) return;
    const int tm = tile / tiles_n, tn = tile % tiles_n;
    const int row0 = tm * 32, col0 = tn * 64;

    // per-lane base pointers; all fragments at immediate offsets from these
    const unsigned short* ap = A  + (size_t)(row0 + lc) * Kd + (hi << 4);
    const unsigned short* bp = Wt + (size_t)(col0 + lc) * Kd + (hi << 4);
    const size_t aoff = (size_t)16 * Kd;          // mt stride (elements)
    const size_t boff = (size_t)16 * Kd;          // nt stride (elements)

    v8f acc[2][4];
#pragma unroll
    for (int i = 0; i < 2; i++)
#pragma unroll
        for (int j = 0; j < 4; j++) acc[i][j] = v8f{0.f,0.f,0.f,0.f,0.f,0.f,0.f,0.f};

    for (int kk = 0; kk < Kd; kk += 32) {
        v16bf a[2], b[4];
#pragma unroll
        for (int mt = 0; mt < 2; mt++)
            a[mt] = load_frag_2x8(ap + mt * aoff, ap + mt * aoff + 8);
#pragma unroll
        for (int nt = 0; nt < 4; nt++)
            b[nt] = *(const v16bf*)(bp + nt * boff);
        __builtin_prefetch(ap + 32, 0, 1);
        __builtin_prefetch(bp + 32, 0, 1);
#pragma unroll
        for (int nt = 0; nt < 4; nt++) {
            acc[0][nt] = __builtin_amdgcn_wmma_f32_16x16x32_bf16(false, a[0], false, b[nt],
                                                                 (short)0, acc[0][nt], false, false);
            acc[1][nt] = __builtin_amdgcn_wmma_f32_16x16x32_bf16(false, a[1], false, b[nt],
                                                                 (short)0, acc[1][nt], false, false);
        }
        ap += 32; bp += 32;
    }
#pragma unroll
    for (int mt = 0; mt < 2; mt++)
#pragma unroll
        for (int nt = 0; nt < 4; nt++) {
            if (TROUT) {
                // lane holds 8 consecutive rows of one column -> one b128 store
                const int col = col0 + 16*nt + lc;
                v8bf pack;
#pragma unroll
                for (int v = 0; v < 8; v++) {
                    float val = acc[mt][nt][v];
                    if (bias) val += bias[col];
                    if (RELU) val = val > 0.f ? val : 0.f;
                    pack[v] = bfraw(f2bf(val));
                }
                *(v8bf*)((unsigned short*)C + (size_t)col * M + row0 + 16*mt + (hi << 3)) = pack;
            } else {
#pragma unroll
                for (int v = 0; v < 8; v++) {
                    const int row = row0 + 16*mt + v + (hi << 3);
                    const int col = col0 + 16*nt + lc;
                    float val = acc[mt][nt][v];
                    if (bias) val += bias[col];
                    if (RELU) val = val > 0.f ? val : 0.f;
                    if (rowscale) val *= rowscale[(size_t)row * rs_stride];
                    out_write(&C[(size_t)row * N + col], val, ACC);
                }
            }
        }
}

// ---------------------------------------------------------------------------
// Streaming (flash-style) attention, all-bf16 operands: one wave per
// (b, h, 16-query block). Per-lane base pointers advanced per key-block;
// K-frags loaded per score tile (short liveness), V-frags overlapped with the
// P LDS round-trip. DPP16 softmax.  Q/K row-major bf16, V transposed bf16.
// ---------------------------------------------------------------------------
__global__ __launch_bounds__(256, 1)
void attn_kernel(const unsigned short* __restrict__ Q,
                 const unsigned short* __restrict__ Km,
                 const unsigned short* __restrict__ Vt,
                 unsigned short* __restrict__ O,
                 const int* __restrict__ amask,
                 int Tq, int Tk, int causal, float scale)
{
    __shared__ unsigned short pbuf[8][16][32];
    const int lane = threadIdx.x & 31;
    const int hi   = lane >> 4;
    const int lc   = lane & 15;
    const int w    = threadIdx.x >> 5;
    const int qblocks = Tq >> 4;
    const int wid = blockIdx.x * (blockDim.x >> 5) + w;
    const int total = Bc * Hc * qblocks;
    if (wid >= total) return;
    const int qb = wid % qblocks;
    const int h  = (wid / qblocks) % Hc;
    const int b  = wid / (qblocks * Hc);
    const size_t Mv = (size_t)Bc * Tk;            // leading dim of Vt rows

    v16bf aq[2];                                  // Q 16x64 as two A fragments
    {
        const unsigned short* qp = Q + (size_t)(b * Tq + qb * 16 + lc) * Dc + h * DHc;
#pragma unroll
        for (int f = 0; f < 2; f++) {
            const unsigned short* p = qp + 32*f + (hi << 4);
            aq[f] = load_frag_2x8(p, p + 8);
        }
    }

    // per-lane base pointers, advanced by fixed strides each key-block
    const unsigned short* kcur = Km + (size_t)(b * Tk + lc) * Dc + h * DHc + (hi << 4);
    const unsigned short* vcur = Vt + (size_t)(h * DHc + lc) * Mv + (size_t)b * Tk + (hi << 4);
    const int* amp = amask ? (amask + b * Tk + lc) : nullptr;
    const size_t kTstep = (size_t)16 * Dc;        // t -> +16 key rows
    const size_t vNstep = (size_t)16 * Mv;        // nt -> +16 Vt rows
    const int qrow_b = qb * 16 + (hi << 3);       // row of acc element v is qrow_b + v

    float mrow[8], lrow[8];
#pragma unroll
    for (int v = 0; v < 8; v++) { mrow[v] = -3.0e38f; lrow[v] = 0.f; }
    v8f o[4];
#pragma unroll
    for (int nt = 0; nt < 4; nt++) o[nt] = v8f{0.f,0.f,0.f,0.f,0.f,0.f,0.f,0.f};

    const int nkb = causal ? (((qb * 16 + 15) >> 5) + 1) : (Tk >> 5);
    for (int kb = 0; kb < nkb; kb++) {
        // ---- scores: per-tile K fragment loads, consumed immediately ----
        v8f s[2];
#pragma unroll
        for (int t = 0; t < 2; t++) {
            const v16bf bk0 = *(const v16bf*)(kcur + (size_t)t * kTstep);
            const v16bf bk1 = *(const v16bf*)(kcur + (size_t)t * kTstep + 32);
            s[t] = v8f{0.f,0.f,0.f,0.f,0.f,0.f,0.f,0.f};
            s[t] = __builtin_amdgcn_wmma_f32_16x16x32_bf16(false, aq[0], false, bk0,
                                                           (short)0, s[t], false, false);
            s[t] = __builtin_amdgcn_wmma_f32_16x16x32_bf16(false, aq[1], false, bk1,
                                                           (short)0, s[t], false, false);
        }
        // ---- scale + mask (value select, no divergence) ----
#pragma unroll
        for (int t = 0; t < 2; t++) {
            const int key = kb * 32 + 16*t + lc;
            float keep = 1.f;
            if (amp) keep = (amp[16*t] != 0) ? 1.f : 0.f;
#pragma unroll
            for (int v = 0; v < 8; v++) {
                const bool ok = (keep != 0.f) && (!causal || key <= qrow_b + v);
                s[t][v] = ok ? s[t][v] * scale : -1.0e9f;
            }
        }
        // ---- online softmax: DPP16 butterflies (rows live in lane halves) ----
        float fac[8];
#pragma unroll
        for (int v = 0; v < 8; v++) {
            const float x = red16_max(fmaxf(s[0][v], s[1][v]));
            const float mnew = fmaxf(mrow[v], x);
            fac[v] = __expf(mrow[v] - mnew);
            mrow[v] = mnew;
            float rs = 0.f;
#pragma unroll
            for (int t = 0; t < 2; t++) {
                const float pv = __expf(s[t][v] - mnew);
                s[t][v] = pv;
                rs += pv;
            }
            rs = red16_sum(rs);
            lrow[v] = lrow[v] * fac[v] + rs;
            o[0][v] *= fac[v]; o[1][v] *= fac[v]; o[2][v] *= fac[v]; o[3][v] *= fac[v];
        }
        // ---- V fragment loads overlap the P LDS round-trip ----
        v16bf bv[4];
#pragma unroll
        for (int nt = 0; nt < 4; nt++)
            bv[nt] = *(const v16bf*)(vcur + (size_t)nt * vNstep);
        // ---- P (D-fragment) -> LDS -> A-fragment, wave-private ----
#pragma unroll
        for (int t = 0; t < 2; t++)
#pragma unroll
            for (int v = 0; v < 8; v++)
                pbuf[w][v + (hi << 3)][16*t + lc] = f2bf(s[t][v]);
        asm volatile("s_wait_dscnt 0x0" ::: "memory");
        v16bf pa;
#pragma unroll
        for (int i = 0; i < 16; i++)
            pa[i] = bfraw(pbuf[w][lc][(hi << 4) + i]);
        // ---- O += P @ V ----
#pragma unroll
        for (int nt = 0; nt < 4; nt++)
            o[nt] = __builtin_amdgcn_wmma_f32_16x16x32_bf16(false, pa, false, bv[nt],
                                                            (short)0, o[nt], false, false);
        kcur += (size_t)32 * Dc;
        vcur += 32;
        if (amp) amp += 32;
    }
#pragma unroll
    for (int v = 0; v < 8; v++) {
        const float inv = 1.f / lrow[v];
        const int row = qb * 16 + v + (hi << 3);
        unsigned short* op = O + (size_t)(b * Tq + row) * Dc + h * DHc;
#pragma unroll
        for (int nt = 0; nt < 4; nt++) op[nt * 16 + lc] = f2bf(o[nt][v] * inv);
    }
}

// ---------------------------------------------------------------------------
// Elementwise / reduction kernels
// ---------------------------------------------------------------------------
__global__ void cvt_kernel(const float* __restrict__ in, unsigned short* __restrict__ out, int n)
{
    int i = blockIdx.x * blockDim.x + threadIdx.x;
    if (i < n) out[i] = f2bf(in[i]);
}

// f32 in[K][N] -> bf16 out[N][K], 32x32 tiles staged in LDS.
__global__ void cvt_t_kernel(const float* __restrict__ in, unsigned short* __restrict__ out,
                             int K, int N)
{
    __shared__ unsigned short tile[32][33];
    const int tx = threadIdx.x & 31;
    const int ty = threadIdx.x >> 5;          // 0..7
    const int k0 = blockIdx.x * 32;
    const int n0 = blockIdx.y * 32;
#pragma unroll
    for (int r = ty; r < 32; r += 8)
        tile[r][tx] = f2bf(in[(size_t)(k0 + r) * N + n0 + tx]);
    __syncthreads();
#pragma unroll
    for (int r = ty; r < 32; r += 8)
        out[(size_t)(n0 + r) * K + k0 + tx] = tile[tx][r];
}

__global__ void embed_kernel(const int* __restrict__ ids, const float* __restrict__ emb,
                             const float* __restrict__ pos, float* __restrict__ y,
                             unsigned short* __restrict__ yb)
{
    int i = blockIdx.x * blockDim.x + threadIdx.x;
    const int n = Bc * Tc * Dc;
    if (i >= n) return;
    const int d  = i & (Dc - 1);
    const int bt = i >> 9;           // Dc == 512
    const int t  = bt % Tc;
    const int id = ids[bt];
    const float v = emb[(size_t)id * Dc + d] + pos[(size_t)t * Dc + d];
    y[i] = v; yb[i] = f2bf(v);
}

// y = LN(a + b) * g + beta ; also writes bf16 copy. One wave per row (D=512).
__global__ void add_ln_kernel(const float* __restrict__ A, const float* __restrict__ Bm,
                              const float* __restrict__ g, const float* __restrict__ bt,
                              float* __restrict__ Y, unsigned short* __restrict__ Yb, int rows)
{
    const int lane = threadIdx.x & 31;
    const int row  = blockIdx.x * (blockDim.x >> 5) + (threadIdx.x >> 5);
    if (row >= rows) return;
    float x[16];
    float s = 0.f;
    const float* ap = A + (size_t)row * Dc;
    const float* bp = Bm + (size_t)row * Dc;
#pragma unroll
    for (int j = 0; j < 16; j++) { const int d = j * 32 + lane; x[j] = ap[d] + bp[d]; s += x[j]; }
    s = red32_sum(s);
    const float mu = s * (1.f / Dc);
    float vv = 0.f;
#pragma unroll
    for (int j = 0; j < 16; j++) { const float t = x[j] - mu; vv += t * t; }
    vv = red32_sum(vv);
    const float rstd = rsqrtf(vv * (1.f / Dc) + 1e-5f);
#pragma unroll
    for (int j = 0; j < 16; j++) {
        const int d = j * 32 + lane;
        const float yv = (x[j] - mu) * rstd * g[d] + bt[d];
        Y[(size_t)row * Dc + d] = yv;
        Yb[(size_t)row * Dc + d] = f2bf(yv);
    }
}

// Router: softmax over E=8, top-2, renormalized gates. One wave per row.
__global__ void router_kernel(const float* __restrict__ Y, const float* __restrict__ rw,
                              const float* __restrict__ rb, float* __restrict__ gates,
                              float* __restrict__ probs, float* __restrict__ sel, int rows)
{
    const int lane = threadIdx.x & 31;
    const int row  = blockIdx.x * (blockDim.x >> 5) + (threadIdx.x >> 5);
    if (row >= rows) return;
    float acc[Ec];
#pragma unroll
    for (int e = 0; e < Ec; e++) acc[e] = 0.f;
    const float* yp = Y + (size_t)row * Dc;
#pragma unroll
    for (int j = 0; j < 16; j++) {
        const int d = j * 32 + lane;
        const float yv = yp[d];
        const float* wp = rw + (size_t)d * Ec;
#pragma unroll
        for (int e = 0; e < Ec; e++) acc[e] += yv * wp[e];
    }
#pragma unroll
    for (int e = 0; e < Ec; e++) acc[e] = red32_sum(acc[e]) + rb[e];
    float mx = acc[0];
#pragma unroll
    for (int e = 1; e < Ec; e++) mx = fmaxf(mx, acc[e]);
    float p[Ec], sum = 0.f;
#pragma unroll
    for (int e = 0; e < Ec; e++) { p[e] = __expf(acc[e] - mx); sum += p[e]; }
    const float inv = 1.f / sum;
#pragma unroll
    for (int e = 0; e < Ec; e++) p[e] *= inv;
    int i1 = 0; float m1 = p[0];
#pragma unroll
    for (int e = 1; e < Ec; e++) if (p[e] > m1) { m1 = p[e]; i1 = e; }
    int i2 = -1; float m2 = -1.f;
#pragma unroll
    for (int e = 0; e < Ec; e++) if (e != i1 && p[e] > m2) { m2 = p[e]; i2 = e; }
    const float denom = m1 + m2;
    if (lane == 0) {
#pragma unroll
        for (int e = 0; e < Ec; e++) {
            const float se = (e == i1 || e == i2) ? 1.f : 0.f;
            probs[(size_t)row * Ec + e] = p[e];
            sel[(size_t)row * Ec + e]   = se;
            gates[(size_t)row * Ec + e] = se * p[e] / denom;
        }
    }
}

// Deterministic load-balance loss: lb += E * sum_e mean(sel)*mean(probs). One block.
__global__ void lb_kernel(const float* __restrict__ probs, const float* __restrict__ sel,
                          float* __restrict__ out_lb, int rows)
{
    __shared__ float sp[256][Ec];
    __shared__ float sf[256][Ec];
    const int tid = threadIdx.x;
    float fp[Ec], pp[Ec];
#pragma unroll
    for (int e = 0; e < Ec; e++) { fp[e] = 0.f; pp[e] = 0.f; }
    for (int r = tid; r < rows; r += 256)
#pragma unroll
        for (int e = 0; e < Ec; e++) {
            fp[e] += sel[(size_t)r * Ec + e];
            pp[e] += probs[(size_t)r * Ec + e];
        }
#pragma unroll
    for (int e = 0; e < Ec; e++) { sf[tid][e] = fp[e]; sp[tid][e] = pp[e]; }
    __syncthreads();
    if (tid == 0) {
        float lb = 0.f;
        for (int e = 0; e < Ec; e++) {
            float fs = 0.f, ps = 0.f;
            for (int i = 0; i < 256; i++) { fs += sf[i][e]; ps += sp[i][e]; }
            lb += (fs / rows) * (ps / rows);
        }
        out_lb[0] += (float)Ec * lb;
    }
}

__global__ void set_scalar_kernel(float* p, float v)
{
    if (blockIdx.x == 0 && threadIdx.x == 0) *p = v;
}

__global__ void copy_kernel(const float* __restrict__ in, float* __restrict__ out, int n)
{
    int i = blockIdx.x * blockDim.x + threadIdx.x;
    if (i < n) out[i] = in[i];
}

// ---------------------------------------------------------------------------
extern "C" void kernel_launch(void* const* d_in, const int* in_sizes, int n_in,
                              void* d_out, int out_size, void* d_ws, size_t ws_size,
                              hipStream_t stream)
{
    (void)in_sizes; (void)n_in; (void)out_size; (void)ws_size;
    const int*   ids  = (const int*)d_in[0];
    const float* enc  = (const float*)d_in[1];
    const int*   am   = (const int*)d_in[2];
    const float* emb  = (const float*)d_in[3];
    const float* pos  = (const float*)d_in[4];
    const float* sa_w = (const float*)d_in[5];
    const float* sa_b = (const float*)d_in[6];
    const float* ca_w = (const float*)d_in[7];
    const float* ca_b = (const float*)d_in[8];
    const float* ln_g = (const float*)d_in[9];
    const float* ln_b = (const float*)d_in[10];
    const float* rw   = (const float*)d_in[11];
    const float* rb   = (const float*)d_in[12];
    const float* w1   = (const float*)d_in[13];
    const float* b1   = (const float*)d_in[14];
    const float* w2   = (const float*)d_in[15];
    const float* b2   = (const float*)d_in[16];
    float* out_y  = (float*)d_out;
    float* out_lb = out_y + (size_t)Bc * Tc * Dc;

    const int BT = Bc * Tc, BS = Bc * Sc;
    char* wp_ = (char*)d_ws;
    auto carve = [&](size_t bytes) -> void* {
        void* r = (void*)wp_;
        wp_ += (bytes + 255) & ~(size_t)255;
        return r;
    };
    float* yf    = (float*)carve((size_t)BT * Dc * 4);
    float* tmpf  = (float*)carve((size_t)BT * Dc * 4);
    float* moef  = (float*)carve((size_t)BT * Dc * 4);
    float* gates = (float*)carve((size_t)BT * Ec * 4);
    float* probs = (float*)carve((size_t)BT * Ec * 4);
    float* sel   = (float*)carve((size_t)BT * Ec * 4);
    unsigned short* xb   = (unsigned short*)carve((size_t)BT * Dc * 2);
    unsigned short* encb = (unsigned short*)carve((size_t)BS * Dc * 2);
    unsigned short* qb   = (unsigned short*)carve((size_t)BT * Dc * 2);
    unsigned short* kb   = (unsigned short*)carve((size_t)BS * Dc * 2);
    unsigned short* vt   = (unsigned short*)carve((size_t)BS * Dc * 2);  // [Dc][B*Tk]
    unsigned short* ab   = (unsigned short*)carve((size_t)BT * Dc * 2);
    unsigned short* wqb  = (unsigned short*)carve((size_t)4 * Dc * Dc * 2); // 4x transposed [Dc][Dc]
    unsigned short* w1b  = (unsigned short*)carve((size_t)Dc * FFc * 2);    // [FFc][Dc]
    unsigned short* w2b  = (unsigned short*)carve((size_t)FFc * Dc * 2);    // [Dc][FFc]
    unsigned short* hb   = (unsigned short*)carve((size_t)BT * FFc * 2);

    auto cvt = [&](const float* src, unsigned short* dst, size_t n) {
        cvt_kernel<<<dim3((unsigned)((n + 255) / 256)), 256, 0, stream>>>(src, dst, (int)n);
    };
    auto cvt_t = [&](const float* src, unsigned short* dst, int K, int N) {
        cvt_t_kernel<<<dim3(K / 32, N / 32), 256, 0, stream>>>(src, dst, K, N);
    };
    auto gemm_f = [&](const unsigned short* A, const unsigned short* Wt, const float* bias,
                      float* C, int M, int N, int Kd) {
        int tiles = (M / 32) * (N / 64);
        gemm_bf16_wmma<false, false, false, float>
            <<<dim3((tiles + 7) / 8), 256, 0, stream>>>(A, Wt, bias, nullptr, 0, C, M, N, Kd);
    };
    auto gemm_b = [&](const unsigned short* A, const unsigned short* Wt, const float* bias,
                      unsigned short* C, int M, int N, int Kd) {
        int tiles = (M / 32) * (N / 64);
        gemm_bf16_wmma<false, false, false, unsigned short>
            <<<dim3((tiles + 7) / 8), 256, 0, stream>>>(A, Wt, bias, nullptr, 0, C, M, N, Kd);
    };
    auto gemm_bt = [&](const unsigned short* A, const unsigned short* Wt, const float* bias,
                       unsigned short* C, int M, int N, int Kd) {   // transposed bf16 out
        int tiles = (M / 32) * (N / 64);
        gemm_bf16_wmma<false, false, true, unsigned short>
            <<<dim3((tiles + 7) / 8), 256, 0, stream>>>(A, Wt, bias, nullptr, 0, C, M, N, Kd);
    };
    auto gemm_relu_b = [&](const unsigned short* A, const unsigned short* Wt, const float* bias,
                           unsigned short* C, int M, int N, int Kd) {
        int tiles = (M / 32) * (N / 64);
        gemm_bf16_wmma<true, false, false, unsigned short>
            <<<dim3((tiles + 7) / 8), 256, 0, stream>>>(A, Wt, bias, nullptr, 0, C, M, N, Kd);
    };
    auto gemm_scaled = [&](const unsigned short* A, const unsigned short* Wt, const float* bias,
                           const float* rs, int rstride, float* C, int M, int N, int Kd, bool acc) {
        int tiles = (M / 32) * (N / 64);
        if (acc)
            gemm_bf16_wmma<false, true, false, float>
                <<<dim3((tiles + 7) / 8), 256, 0, stream>>>(A, Wt, bias, rs, rstride, C, M, N, Kd);
        else
            gemm_bf16_wmma<false, false, false, float>
                <<<dim3((tiles + 7) / 8), 256, 0, stream>>>(A, Wt, bias, rs, rstride, C, M, N, Kd);
    };

    set_scalar_kernel<<<1, 32, 0, stream>>>(out_lb, 0.f);
    embed_kernel<<<dim3(BT * Dc / 256), 256, 0, stream>>>(ids, emb, pos, yf, xb);
    cvt(enc, encb, (size_t)BS * Dc);

    const float scale = 0.125f;   // 1/sqrt(DH)
    const int attn_blocks = Bc * Hc * (Tc / 16) / 8;

    for (int l = 0; l < Lc; l++) {
        // ---- self attention (q/k bf16 row-major, v bf16 transposed) ----
        for (int j = 0; j < 4; j++)
            cvt_t(sa_w + ((size_t)(l * 4 + j)) * Dc * Dc, wqb + (size_t)j * Dc * Dc, Dc, Dc);
        const float* sb = sa_b + (size_t)l * 4 * Dc;
        gemm_b (xb, wqb + 0 * (size_t)Dc * Dc, sb + 0 * Dc, qb, BT, Dc, Dc);
        gemm_b (xb, wqb + 1 * (size_t)Dc * Dc, sb + 1 * Dc, kb, BT, Dc, Dc);
        gemm_bt(xb, wqb + 2 * (size_t)Dc * Dc, sb + 2 * Dc, vt, BT, Dc, Dc);
        attn_kernel<<<dim3(attn_blocks), 256, 0, stream>>>(qb, kb, vt, ab, am, Tc, Tc, 1, scale);
        gemm_f(ab, wqb + 3 * (size_t)Dc * Dc, sb + 3 * Dc, tmpf, BT, Dc, Dc);
        add_ln_kernel<<<dim3(BT / 8), 256, 0, stream>>>(
            yf, tmpf, ln_g + ((size_t)l * 3 + 0) * Dc, ln_b + ((size_t)l * 3 + 0) * Dc, yf, xb, BT);

        // ---- cross attention ----
        for (int j = 0; j < 4; j++)
            cvt_t(ca_w + ((size_t)(l * 4 + j)) * Dc * Dc, wqb + (size_t)j * Dc * Dc, Dc, Dc);
        const float* cb = ca_b + (size_t)l * 4 * Dc;
        gemm_b (xb,   wqb + 0 * (size_t)Dc * Dc, cb + 0 * Dc, qb, BT, Dc, Dc);
        gemm_b (encb, wqb + 1 * (size_t)Dc * Dc, cb + 1 * Dc, kb, BS, Dc, Dc);
        gemm_bt(encb, wqb + 2 * (size_t)Dc * Dc, cb + 2 * Dc, vt, BS, Dc, Dc);
        attn_kernel<<<dim3(attn_blocks), 256, 0, stream>>>(qb, kb, vt, ab, nullptr, Tc, Sc, 0, scale);
        gemm_f(ab, wqb + 3 * (size_t)Dc * Dc, cb + 3 * Dc, tmpf, BT, Dc, Dc);
        add_ln_kernel<<<dim3(BT / 8), 256, 0, stream>>>(
            yf, tmpf, ln_g + ((size_t)l * 3 + 1) * Dc, ln_b + ((size_t)l * 3 + 1) * Dc, yf, xb, BT);

        // ---- sparse MoE (dense expert compute, gated accumulate) ----
        router_kernel<<<dim3(BT / 8), 256, 0, stream>>>(
            yf, rw + (size_t)l * Dc * Ec, rb + (size_t)l * Ec, gates, probs, sel, BT);
        lb_kernel<<<1, 256, 0, stream>>>(probs, sel, out_lb, BT);
        for (int e = 0; e < Ec; e++) {
            cvt_t(w1 + ((size_t)(l * Ec + e)) * Dc * FFc, w1b, Dc, FFc);   // -> [FFc][Dc]
            cvt_t(w2 + ((size_t)(l * Ec + e)) * FFc * Dc, w2b, FFc, Dc);   // -> [Dc][FFc]
            gemm_relu_b(xb, w1b, b1 + ((size_t)(l * Ec + e)) * FFc, hb, BT, FFc, Dc);
            gemm_scaled(hb, w2b, b2 + ((size_t)(l * Ec + e)) * Dc,
                        gates + e, Ec, moef, BT, Dc, FFc, e > 0);
        }
        add_ln_kernel<<<dim3(BT / 8), 256, 0, stream>>>(
            yf, moef, ln_g + ((size_t)l * 3 + 2) * Dc, ln_b + ((size_t)l * 3 + 2) * Dc, yf, xb, BT);
    }

    copy_kernel<<<dim3(BT * Dc / 256), 256, 0, stream>>>(yf, out_y, BT * Dc);
}
